// FlaGdnpControlConditionedSequenceMixer_47064251630311
// MI455X (gfx1250) — matrix operand order repaired and checked by reading, not compiled, not run on hardware
//
#include <hip/hip_runtime.h>

// ---------------------------------------------------------------------------
// FlaGdnpControlConditionedSequenceMixer for MI455X (gfx1250, wave32, WMMA)
// B=2 S=1024 D=1024 H=16 DH=64 BW=128 K=4
// ---------------------------------------------------------------------------

typedef __attribute__((ext_vector_type(16))) _Float16 v16h;
typedef __attribute__((ext_vector_type(2)))  _Float16 h2;
typedef __attribute__((ext_vector_type(8)))  float    v8f;

#define BB   2
#define SS   1024
#define DD   1024
#define HH   16
#define DHH  64
#define BWC  128
#define KK   4
#define NDELTA (3*DD + HH)          // 3088
#define NCTRL  (3*DD + 2*HH + DD)   // 4128
#define EPSF 1e-6f

// GEMM macro-tile geometry: 256 threads = 8 waves, 2x4 wave grid,
// each wave computes 32x32 (2x2 WMMA 16x16x32 subtiles).
#define MT 64
#define NT 128
#define KT 32
#define AS_PAD 2                    // LDS bank stagger (halves)
#define BSP_PAD 4                   // LDS bank stagger (pair-words)

__device__ __forceinline__ float sigmoidf_(float x) { return 1.0f / (1.0f + __expf(-x)); }
__device__ __forceinline__ float siluf_(float x)    { return x * sigmoidf_(x); }

// ---------------------------------------------------------------------------
// LDS fragment loaders, wave32; A and B share one contiguous k order (not the ISA table order).
// A (16-bit, 16x32): lane L holds row M=L%16;
//   VGPR j holds K = (L>=16?16:0) + 2j  (and +1)
//   -> pairs are row-adjacent in As, compiler fuses to ds_load_2addr_b32.
// B (16-bit, 32x16): lane L holds col N=L%16; K = (L>=16?16:0) + 2j (and +1)
//   -> Bs is PAIR-INTERLEAVED: element (k,n) in half (k&1) of word
//      Bs[k>>1][n], so each lane pair is one aligned ds_load_b32.
// ---------------------------------------------------------------------------
__device__ __forceinline__ v16h load_a_frag(const _Float16 (*As)[KT + AS_PAD],
                                            int mloc, bool hi) {
  v16h a;
#pragma unroll
  for (int j = 0; j < 8; ++j) {
    int kk = (hi ? 16 : 0) + (j << 1);
    a[2 * j]     = As[mloc][kk];
    a[2 * j + 1] = As[mloc][kk + 1];
  }
  return a;
}

__device__ __forceinline__ v16h load_b_frag(const _Float16 (*Bs)[NT + BSP_PAD][2],
                                            int nloc, bool hi) {
  v16h b;
#pragma unroll
  for (int j = 0; j < 8; ++j) {
    int kp = (hi ? 8 : 0) + j;                 // pair index: kk = 2*kp
    const h2 p = *(const h2*)&Bs[kp][nloc][0]; // one 4B-aligned LDS dword
    b[2 * j]     = p[0];
    b[2 * j + 1] = p[1];
  }
  return b;
}

// ---------------------------------------------------------------------------
// Tiled WMMA GEMM: C[M,N] = A[M,K] * B[K,N] (+bias[N]); optional B^T
// (TRANSB: B element (k,n) = Bm[n*K + k]).  f32 in/out, f16 multiply,
// f32 accumulate.  Requires M % 64 == 0, K % 32 == 0; N % 16 == 0 (edge
// tiles guarded).  Grid: ((N+127)/128, M/64), block 256.
// ---------------------------------------------------------------------------
template<bool TRANSB, bool HASBIAS>
__global__ __launch_bounds__(256)
void wmma_gemm_tiled(const float* __restrict__ A,
                     const float* __restrict__ Bm,
                     const float* __restrict__ bias,
                     float* __restrict__ C,
                     int M, int N, int K) {
  __shared__ _Float16 As[MT][KT + AS_PAD];           // [m][k]
  __shared__ _Float16 Bs[KT / 2][NT + BSP_PAD][2];   // [k/2][n][k&1]

  const int tid   = threadIdx.x;
  const int wave  = tid >> 5;
  const int lane  = tid & 31;
  const int rlo   = lane & 15;
  const bool hi   = lane >= 16;
  const int wm    = wave >> 2;               // 0..1
  const int wn    = wave & 3;                // 0..3

  const int nbase = blockIdx.x * NT;
  const int mbase = blockIdx.y * MT;

  v8f acc[2][2] = {};

  for (int k0 = 0; k0 < K; k0 += KT) {
    // ---- stage A tile: 64x32 f32 -> f16, 128-bit coalesced loads ----
#pragma unroll
    for (int it = 0; it < 2; ++it) {
      int chunk = tid + it * 256;            // 0..511
      int row = chunk >> 3;
      int kc  = (chunk & 7) << 2;
      const float4 av =
          *(const float4*)(A + (size_t)(mbase + row) * K + k0 + kc);
      As[row][kc + 0] = (_Float16)av.x;
      As[row][kc + 1] = (_Float16)av.y;
      As[row][kc + 2] = (_Float16)av.z;
      As[row][kc + 3] = (_Float16)av.w;
    }
    // ---- stage B tile: 32x128 f16, pair-interleaved [k/2][n][2] ----
    if (TRANSB) {
#pragma unroll
      for (int it = 0; it < 4; ++it) {
        int chunk = tid + it * 256;          // 0..1023
        int nr = chunk >> 3;                 // 0..127
        int kc = (chunk & 7) << 2;           // 0..28 (multiple of 4)
        int n  = nbase + nr;
        float4 bv = make_float4(0.f, 0.f, 0.f, 0.f);
        if (n < N) bv = *(const float4*)(Bm + (size_t)n * K + k0 + kc);
        h2 w0, w1;
        w0[0] = (_Float16)bv.x; w0[1] = (_Float16)bv.y;   // k = kc, kc+1
        w1[0] = (_Float16)bv.z; w1[1] = (_Float16)bv.w;   // k = kc+2, kc+3
        *(h2*)&Bs[(kc >> 1) + 0][nr][0] = w0;
        *(h2*)&Bs[(kc >> 1) + 1][nr][0] = w1;
      }
    } else {
#pragma unroll
      for (int it = 0; it < 4; ++it) {
        int chunk = tid + it * 256;          // 0..1023
        int kp = chunk >> 6;                 // 0..15 (pair of k rows)
        int np = (chunk & 63) << 1;          // 0..126
        int n  = nbase + np;
        float2 r0 = make_float2(0.f, 0.f);
        float2 r1 = make_float2(0.f, 0.f);
        if (n < N) {
          r0 = *(const float2*)(Bm + (size_t)(k0 + 2 * kp) * N + n);
          r1 = *(const float2*)(Bm + (size_t)(k0 + 2 * kp + 1) * N + n);
        }
        h2 w0, w1;
        w0[0] = (_Float16)r0.x; w0[1] = (_Float16)r1.x;   // col np
        w1[0] = (_Float16)r0.y; w1[1] = (_Float16)r1.y;   // col np+1
        *(h2*)&Bs[kp][np][0]     = w0;      // adjacent -> fusable to b64
        *(h2*)&Bs[kp][np + 1][0] = w1;
      }
    }
    __syncthreads();

    // ---- fragments + 2x2 WMMA ----
    const v16h a0 = load_a_frag(As, wm * 32 + rlo, hi);
    const v16h a1 = load_a_frag(As, wm * 32 + 16 + rlo, hi);
    const v16h b0 = load_b_frag(Bs, wn * 32 + rlo, hi);
    const v16h b1 = load_b_frag(Bs, wn * 32 + 16 + rlo, hi);

    acc[0][0] = __builtin_amdgcn_wmma_f32_16x16x32_f16(
        false, a0, false, b0, (short)0, acc[0][0], false, false);
    acc[0][1] = __builtin_amdgcn_wmma_f32_16x16x32_f16(
        false, a0, false, b1, (short)0, acc[0][1], false, false);
    acc[1][0] = __builtin_amdgcn_wmma_f32_16x16x32_f16(
        false, a1, false, b0, (short)0, acc[1][0], false, false);
    acc[1][1] = __builtin_amdgcn_wmma_f32_16x16x32_f16(
        false, a1, false, b1, (short)0, acc[1][1], false, false);
    __syncthreads();
  }

  // ---- epilogue: C layout VGPR r -> (M = r + (hi?8:0), N = lane%16) ----
#pragma unroll
  for (int i = 0; i < 2; ++i) {
#pragma unroll
    for (int j = 0; j < 2; ++j) {
      int ncol = nbase + wn * 32 + j * 16 + rlo;
      if (ncol >= N) continue;
      float bv = HASBIAS ? bias[ncol] : 0.0f;
#pragma unroll
      for (int r = 0; r < 8; ++r) {
        int row = mbase + wm * 32 + i * 16 + r + (hi ? 8 : 0);
        C[(size_t)row * N + ncol] = acc[i][j][r] + bv;
      }
    }
  }
}

// ---------------------------------------------------------------------------
// Rotary-state scan over the bottleneck: 128 independent chains (b, pair j).
// ---------------------------------------------------------------------------
__global__ void rot_scan_kernel(const float* __restrict__ p,
                                const float* __restrict__ theta,
                                float* __restrict__ states) {
  const int idx  = blockIdx.x * blockDim.x + threadIdx.x;
  const int half = BWC / 2;
  if (idx >= BB * half) return;
  const int b = idx / half;
  const int j = idx % half;
  const float c  = __cosf(theta[j]);
  const float sn = __sinf(theta[j]);
  float s0 = 0.0f, s1 = 0.0f;
  for (int t = 0; t < SS; ++t) {
    const size_t base = ((size_t)b * SS + t) * BWC;
    const float x0 = p[base + 2 * j];
    const float x1 = p[base + 2 * j + 1];
    const float n0 = c * s0 - sn * s1 + x0;
    const float n1 = sn * s0 + c * s1 + x1;
    s0 = n0; s1 = n1;
    states[base + 2 * j]     = s0;
    states[base + 2 * j + 1] = s1;
  }
}

// ---------------------------------------------------------------------------
// RMSNorm over BW=128 per row (one block of 128 threads per (b,t)).
// ---------------------------------------------------------------------------
__global__ void rmsnorm_bw_kernel(const float* __restrict__ x,
                                  const float* __restrict__ w,
                                  float* __restrict__ y) {
  __shared__ float red[BWC];
  const int row = blockIdx.x;
  const int tid = threadIdx.x;
  const float v = x[(size_t)row * BWC + tid];
  red[tid] = v * v;
  __syncthreads();
  for (int s = BWC / 2; s > 0; s >>= 1) {
    if (tid < s) red[tid] += red[tid + s];
    __syncthreads();
  }
  const float inv = rsqrtf(red[0] * (1.0f / BWC) + EPSF);
  y[(size_t)row * BWC + tid] = v * inv * w[tid];
}

// ---------------------------------------------------------------------------
// Fused: qkv pre-act = ctrl + ramp*delta -> causal depthwise conv(K=4) ->
// SiLU -> per-head l2norm (q,k) ; also beta = sigmoid(bet_l + ramp_b*bd),
// g = sigmoid(dec_l).  One 64-thread block per (b,t,h).
// ---------------------------------------------------------------------------
__global__ void qkv_conv_kernel(const float* __restrict__ ctrl,
                                const float* __restrict__ deltas,
                                const float* __restrict__ qkv_ramp,   // [3,D]
                                const float* __restrict__ beta_ramp,  // [H]
                                const float* __restrict__ qw,         // [D,K]
                                const float* __restrict__ kw,
                                const float* __restrict__ vw,
                                float* __restrict__ qo,
                                float* __restrict__ ko,
                                float* __restrict__ vo,
                                float* __restrict__ betao,
                                float* __restrict__ go) {
  const int bth = blockIdx.x;               // b*S*H + t*H + h
  const int h  = bth % HH;
  const int bt = bth / HH;
  const int t  = bt % SS;
  const int b  = bt / SS;
  const int lane = threadIdx.x;             // 0..63
  const int c = h * DHH + lane;             // channel in [0,D)

  const float rq = sigmoidf_(qkv_ramp[0 * DD + c]);
  const float rk = sigmoidf_(qkv_ramp[1 * DD + c]);
  const float rv = sigmoidf_(qkv_ramp[2 * DD + c]);

  float aq = 0.0f, ak = 0.0f, av = 0.0f;
#pragma unroll
  for (int i = 0; i < KK; ++i) {
    const int tt = t - (KK - 1) + i;
    if (tt < 0) continue;
    const size_t cb = ((size_t)b * SS + tt) * NCTRL;
    const size_t db = ((size_t)b * SS + tt) * NDELTA;
    const float xq = ctrl[cb + c]          + deltas[db + c]          * rq;
    const float xk = ctrl[cb + DD + c]     + deltas[db + DD + c]     * rk;
    const float xv = ctrl[cb + 2 * DD + c] + deltas[db + 2 * DD + c] * rv;
    aq += xq * qw[c * KK + i];
    ak += xk * kw[c * KK + i];
    av += xv * vw[c * KK + i];
  }
  float q = siluf_(aq), k = siluf_(ak), v = siluf_(av);

  __shared__ float rq2[DHH], rk2[DHH];
  rq2[lane] = q * q;
  rk2[lane] = k * k;
  __syncthreads();
  for (int s = DHH / 2; s > 0; s >>= 1) {
    if (lane < s) { rq2[lane] += rq2[lane + s]; rk2[lane] += rk2[lane + s]; }
    __syncthreads();
  }
  q *= rsqrtf(rq2[0] + EPSF);               // l2norm: sum, not mean
  k *= rsqrtf(rk2[0] + EPSF);

  const size_t ob = ((size_t)b * SS + t) * DD + c;
  qo[ob] = q; ko[ob] = k; vo[ob] = v;

  if (lane == 0) {
    const size_t cb = ((size_t)b * SS + t) * NCTRL;
    const size_t db = ((size_t)b * SS + t) * NDELTA;
    const float dec = ctrl[cb + 3 * DD + h];
    const float bet = ctrl[cb + 3 * DD + HH + h]
                    + deltas[db + 3 * DD + h] * sigmoidf_(beta_ramp[h]);
    const size_t ib = ((size_t)b * SS + t) * HH + h;
    go[ib]    = sigmoidf_(dec);
    betao[ib] = sigmoidf_(bet);
  }
}

// ---------------------------------------------------------------------------
// Gated delta-rule recurrence. One 64-thread block per (b,h); each thread
// owns one state column (64 f32 in registers). k/q broadcast via LDS.
// ---------------------------------------------------------------------------
__global__ void delta_rule_scan_kernel(const float* __restrict__ q,
                                       const float* __restrict__ k,
                                       const float* __restrict__ v,
                                       const float* __restrict__ beta,
                                       const float* __restrict__ g,
                                       float* __restrict__ o) {
  const int bh = blockIdx.x;
  const int h = bh % HH;
  const int b = bh / HH;
  const int col = threadIdx.x;              // output (value) dim 0..63

  float St[DHH];
#pragma unroll
  for (int j = 0; j < DHH; ++j) St[j] = 0.0f;

  __shared__ float lk[DHH], lq[DHH];
  for (int t = 0; t < SS; ++t) {
    const size_t base = (((size_t)b * SS + t) * HH + h) * DHH;
    lk[col] = k[base + col];
    lq[col] = q[base + col];
    __syncthreads();
    const size_t sb = ((size_t)b * SS + t) * HH + h;
    const float gt = g[sb];
    const float bt = beta[sb];
    const float vt = v[base + col];

    float kS = 0.0f;
#pragma unroll
    for (int j = 0; j < DHH; ++j) {         // decay + k·S
      St[j] *= gt;
      kS = fmaf(lk[j], St[j], kS);
    }
    const float upd = bt * (vt - kS);
    float ov = 0.0f;
#pragma unroll
    for (int j = 0; j < DHH; ++j) {         // rank-1 update + q·S
      St[j] = fmaf(lk[j], upd, St[j]);
      ov = fmaf(lq[j], St[j], ov);
    }
    o[base + col] = ov;
    __syncthreads();                        // protect LDS reuse next step
  }
}

// ---------------------------------------------------------------------------
// Gated per-head RMSNorm * SiLU(gate). One 64-thread block per (b,t,h).
// ---------------------------------------------------------------------------
__global__ void gated_norm_kernel(const float* __restrict__ o,
                                  const float* __restrict__ ctrl,
                                  const float* __restrict__ out_norm_w,
                                  float* __restrict__ y) {
  const int bth = blockIdx.x;
  const int h  = bth % HH;
  const int bt = bth / HH;
  const int t  = bt % SS;
  const int b  = bt / SS;
  const int lane = threadIdx.x;

  const size_t base = ((size_t)b * SS + t) * DD + h * DHH + lane;
  const float ov = o[base];

  __shared__ float red[DHH];
  red[lane] = ov * ov;
  __syncthreads();
  for (int s = DHH / 2; s > 0; s >>= 1) {
    if (lane < s) red[lane] += red[lane + s];
    __syncthreads();
  }
  const float n = ov * rsqrtf(red[0] * (1.0f / DHH) + EPSF) * out_norm_w[lane];
  const float gate = ctrl[((size_t)b * SS + t) * NCTRL + 3 * DD + 2 * HH + h * DHH + lane];
  y[base] = n * siluf_(gate);
}

// ---------------------------------------------------------------------------
// Host-side orchestration
// ---------------------------------------------------------------------------
extern "C" void kernel_launch(void* const* d_in, const int* in_sizes, int n_in,
                              void* d_out, int out_size, void* d_ws, size_t ws_size,
                              hipStream_t stream) {
  const float* hidden        = (const float*)d_in[0];   // [B,S,D]
  const float* cond_in_W     = (const float*)d_in[1];   // [D,BW]
  const float* rot_theta     = (const float*)d_in[2];   // [BW/2]
  const float* cond_norm_w   = (const float*)d_in[3];   // [BW]
  const float* delta_W       = (const float*)d_in[4];   // [BW,3D+H]
  const float* qkv_ramp      = (const float*)d_in[5];   // [3,D]
  const float* beta_ramp     = (const float*)d_in[6];   // [H]
  const float* ctrl_W        = (const float*)d_in[7];   // [D,NCTRL]
  const float* ctrl_b        = (const float*)d_in[8];   // [NCTRL]
  const float* q_conv_w      = (const float*)d_in[9];   // [D,K]
  const float* k_conv_w      = (const float*)d_in[10];
  const float* v_conv_w      = (const float*)d_in[11];
  const float* out_norm_w    = (const float*)d_in[12];  // [DH]
  const float* out_proj_W    = (const float*)d_in[13];  // [D,D]
  const float* out_proj_b    = (const float*)d_in[14];  // [D]
  float* out = (float*)d_out;

  const int M = BB * SS;                                 // 2048 rows

  // workspace carve-out
  char* ws = (char*)d_ws;
  size_t off = 0;
  auto alloc = [&](size_t bytes) -> float* {
    float* p = (float*)(ws + off);
    off = (off + bytes + 255) & ~(size_t)255;
    return p;
  };
  float* p_buf      = alloc((size_t)M * BWC * 4);
  float* states_buf = alloc((size_t)M * BWC * 4);
  float* cond_buf   = alloc((size_t)M * BWC * 4);
  float* deltas_buf = alloc((size_t)M * NDELTA * 4);
  float* ctrl_buf   = alloc((size_t)M * NCTRL * 4);
  float* q_buf      = alloc((size_t)M * DD * 4);
  float* k_buf      = alloc((size_t)M * DD * 4);
  float* v_buf      = alloc((size_t)M * DD * 4);
  float* beta_buf   = alloc((size_t)M * HH * 4);
  float* g_buf      = alloc((size_t)M * HH * 4);
  float* o_buf      = alloc((size_t)M * DD * 4);
  float* y_buf      = alloc((size_t)M * DD * 4);
  (void)ws_size; (void)in_sizes; (void)n_in; (void)out_size;

  auto ggrid = [](int N) { return dim3((unsigned)((N + NT - 1) / NT), (unsigned)(M / MT)); };

  // 1) p = hidden @ cond_in_W                      [2048 x 1024] x [1024 x 128]
  wmma_gemm_tiled<false, false><<<ggrid(BWC), 256, 0, stream>>>(
      hidden, cond_in_W, nullptr, p_buf, M, BWC, DD);

  // 2) rotary-state scan over S (128 independent chains)
  rot_scan_kernel<<<1, 128, 0, stream>>>(p_buf, rot_theta, states_buf);

  // 3) cond = RMSNorm(states) * w
  rmsnorm_bw_kernel<<<M, BWC, 0, stream>>>(states_buf, cond_norm_w, cond_buf);

  // 4) deltas = cond @ delta_W                     [2048 x 128] x [128 x 3088]
  wmma_gemm_tiled<false, false><<<ggrid(NDELTA), 256, 0, stream>>>(
      cond_buf, delta_W, nullptr, deltas_buf, M, NDELTA, BWC);

  // 5) ctrl = hidden @ ctrl_W + ctrl_b             [2048 x 1024] x [1024 x 4128]
  wmma_gemm_tiled<false, true><<<ggrid(NCTRL), 256, 0, stream>>>(
      hidden, ctrl_W, ctrl_b, ctrl_buf, M, NCTRL, DD);

  // 6) fused causal dwconv + SiLU + l2norm + beta/gate extraction
  qkv_conv_kernel<<<BB * SS * HH, DHH, 0, stream>>>(
      ctrl_buf, deltas_buf, qkv_ramp, beta_ramp,
      q_conv_w, k_conv_w, v_conv_w,
      q_buf, k_buf, v_buf, beta_buf, g_buf);

  // 7) gated delta-rule recurrence (32 chains, state in registers)
  delta_rule_scan_kernel<<<BB * HH, DHH, 0, stream>>>(
      q_buf, k_buf, v_buf, beta_buf, g_buf, o_buf);

  // 8) gated per-head RMSNorm * SiLU(gate)
  gated_norm_kernel<<<BB * SS * HH, DHH, 0, stream>>>(
      o_buf, ctrl_buf, out_norm_w, y_buf);

  // 9) out = y @ out_proj_W^T + out_proj_b         [2048 x 1024] x [1024 x 1024]^T
  wmma_gemm_tiled<true, true><<<ggrid(DD), 256, 0, stream>>>(
      y_buf, out_proj_W, out_proj_b, out, M, DD, DD);
}
